// SSMLayer_51161650430323
// MI455X (gfx1250) — compile-verified, matrix operand order checked by reading, not run on hardware
//
#include <hip/hip_runtime.h>
#include <hip/hip_bf16.h>

// ---------------------------------------------------------------------------
// SSM layer: out = silu(causal_dwconv(glu(x @ W_in))) @ W_out
// B=4, L=4096, D=1024, INNER=2048, K=4.  BF16 WMMA, fp32 accumulate.
// Pipeline:
//   prep:   x -> bf16 copy; W_in, W_out -> bf16, transposed to K-contiguous
//   gemm1:  TDM-fed double-buffered BF16 WMMA + fused GLU -> bf16 workspace
//   gemm2:  fused causal dwconv+SiLU A-producer, TDM-fed B, WMMA -> fp32 out
// ---------------------------------------------------------------------------

#define B_      4
#define L_      4096
#define D_      1024
#define INNER_  2048
#define M_      (B_ * L_)      // 16384 rows
#define KCONV   4

typedef __attribute__((ext_vector_type(16))) __bf16 v16bf;
typedef __attribute__((ext_vector_type(8)))  float  v8f;
typedef unsigned int u32x4 __attribute__((ext_vector_type(4)));
typedef int          i32x4 __attribute__((ext_vector_type(4)));
typedef int          i32x8 __attribute__((ext_vector_type(8)));

#if defined(__gfx1250__) && __has_builtin(__builtin_amdgcn_tensor_load_to_lds)
#define HAVE_TDM 1
#else
#define HAVE_TDM 0
#endif

__device__ __forceinline__ unsigned f2bf(float x) {
    unsigned u = __builtin_bit_cast(unsigned, x);
    u += 0x7FFFu + ((u >> 16) & 1u);          // round-to-nearest-even
    return u >> 16;
}
__device__ __forceinline__ float bf2f(unsigned h16) {
    return __builtin_bit_cast(float, h16 << 16);   // caller pre-masks low 16 bits
}

// ---------------------------------------------------------------------------
// TDM: issue a 2D tile DMA (bf16 elements) global -> LDS.
// D# per CDNA5 ISA ch.8: group0 = count/lds_addr/global_addr/type=2,
// group1 = data_size=2B, tensor dims (==tile dims: tiles always in-bounds),
// tile dims, dim0 stride.  Groups 2/3 zero (2-D).  Tracked by TENSORcnt.
// ---------------------------------------------------------------------------
__device__ __forceinline__ void tdm_load_2d(unsigned lds_byte_addr, const void* gptr,
                                            unsigned tile_w, unsigned tile_h,
                                            unsigned row_stride_elems) {
#if HAVE_TDM
    const unsigned long long ga = (unsigned long long)gptr;
    u32x4 g0;
    g0.x = 1u;                                               // count=1, user mode
    g0.y = lds_byte_addr;                                    // [63:32] lds_addr
    g0.z = (unsigned)(ga & 0xFFFFFFFFull);                   // global_addr lo
    g0.w = (unsigned)((ga >> 32) & 0x01FFFFFFull) | (2u << 30); // addr hi | type=2
    i32x8 g1;
    g1[0] = (int)(1u << 16);                                 // data_size=1 (2 bytes)
    g1[1] = (int)((tile_w & 0xFFFFu) << 16);                 // tensor_dim0[15:0]
    g1[2] = (int)(((tile_w >> 16) & 0xFFFFu) | ((tile_h & 0xFFFFu) << 16)); // td0 hi|td1 lo
    g1[3] = (int)(((tile_h >> 16) & 0xFFFFu) | ((tile_w & 0xFFFFu) << 16)); // td1 hi|tile_dim0
    g1[4] = (int)(tile_h & 0xFFFFu);                         // tile_dim1 (tile_dim2=0)
    g1[5] = (int)row_stride_elems;                           // tensor_dim0_stride lo
    g1[6] = 0;
    g1[7] = 0;
    const i32x4 z4 = {0, 0, 0, 0};
#if __clang_major__ >= 23
    const i32x8 z8 = {0, 0, 0, 0, 0, 0, 0, 0};
    __builtin_amdgcn_tensor_load_to_lds(g0, g1, z4, z4, z8, 0);
#else
    __builtin_amdgcn_tensor_load_to_lds(g0, g1, z4, z4, 0);
#endif
#else
    (void)lds_byte_addr; (void)gptr; (void)tile_w; (void)tile_h; (void)row_stride_elems;
#endif
}

__device__ __forceinline__ unsigned lds_addr_of(const void* p) {
    return (unsigned)(unsigned long long)p;   // shared aperture: addr[31:0] = LDS offset
}

// Fallback cooperative copy: rows x 32 bf16, K-contiguous source rows.
__device__ __forceinline__ void stage_tile_bf16(unsigned short* dst, const unsigned short* src,
                                                int rows, int strideElems, int tid) {
    const int nseg = rows * 4;                 // uint4 = 8 bf16 -> 4 segs/row
    for (int s = tid; s < nseg; s += 256) {
        const int r = s >> 2, c = (s & 3) << 3;
        *(uint4*)&dst[r * 32 + c] = *(const uint4*)&src[(size_t)r * strideElems + c];
    }
}

// Load a 16x32 bf16 operand fragment from K-contiguous LDS (stride 32 elems).
//   lanes 0-15 : row = lane,    VGPR0-3 = K 0..7,   VGPR4-7 = K 16..23
//   lanes16-31 : row = lane-16, VGPR0-3 = K 8..15,  VGPR4-7 = K 24..31
__device__ __forceinline__ v16bf load_frag(const unsigned short* base, int lane) {
    const int r  = lane & 15;
    const int kh = (lane >> 4) << 3;
    union { uint4 q[2]; v16bf v; } u;
    u.q[0] = *(const uint4*)(base + r * 32 + kh);
    u.q[1] = *(const uint4*)(base + r * 32 + kh + 16);
    return u.v;
}

// ---------------------------------------------------------------------------
// Preprocessing kernels
// ---------------------------------------------------------------------------
__global__ __launch_bounds__(256) void cvt_bf16_kernel(const float* __restrict__ src,
                                                       unsigned short* __restrict__ dst,
                                                       int nseg8) {
    const int s = blockIdx.x * 256 + threadIdx.x;
    if (s < nseg8) {
        const float4 a = *(const float4*)&src[(size_t)s * 8];
        const float4 b = *(const float4*)&src[(size_t)s * 8 + 4];
        *(uint4*)&dst[(size_t)s * 8] = make_uint4(
            f2bf(a.x) | (f2bf(a.y) << 16), f2bf(a.z) | (f2bf(a.w) << 16),
            f2bf(b.x) | (f2bf(b.y) << 16), f2bf(b.z) | (f2bf(b.w) << 16));
    }
}

// dst[c][r] = bf16(src[r][c]);  src: [R][C] fp32, dst: [C][R] bf16.  64x64 LDS tiles.
__global__ __launch_bounds__(256) void transpose_bf16_kernel(const float* __restrict__ src,
                                                             unsigned short* __restrict__ dst,
                                                             int R, int C) {
    __shared__ unsigned short t[64][72];       // 72: 16B-aligned rows, bank-spread
    const int r0 = blockIdx.y * 64, c0 = blockIdx.x * 64;
#pragma unroll
    for (int i = 0; i < 4; ++i) {
        const int s = threadIdx.x + i * 256;   // 1024 float4 segs
        const int r = s >> 4, c4 = (s & 15) << 2;
        const float4 v = *(const float4*)&src[(size_t)(r0 + r) * C + c0 + c4];
        t[c4 + 0][r] = (unsigned short)f2bf(v.x);
        t[c4 + 1][r] = (unsigned short)f2bf(v.y);
        t[c4 + 2][r] = (unsigned short)f2bf(v.z);
        t[c4 + 3][r] = (unsigned short)f2bf(v.w);
    }
    __syncthreads();
#pragma unroll
    for (int i = 0; i < 2; ++i) {
        const int s = threadIdx.x + i * 256;   // 512 uint4 segs
        const int c = s >> 3, r8 = (s & 7) << 3;
        *(uint4*)&dst[(size_t)(c0 + c) * R + r0 + r8] = *(const uint4*)&t[c][r8];
    }
}

// ---------------------------------------------------------------------------
// Kernel 1: h_gated[m,n] = (x@W_in)[:,n] * sigmoid((x@W_in)[:,n+2048])
// Block 128 rows x 64 gated cols (both GLU halves), K-step 32, double-buffered
// TDM staging.  8 waves 4(M)x2(N); wave = 32x32 gated = 2x2 tiles per half.
// ---------------------------------------------------------------------------
__global__ __launch_bounds__(256) void gemm1_glu_kernel(
        const unsigned short* __restrict__ xbf,   // [M_, D_] bf16
        const unsigned short* __restrict__ WinT,  // [2*INNER_, D_] bf16 (n-major)
        unsigned short* __restrict__ hg) {        // [M_, INNER_] bf16
    __shared__ unsigned short sA[2][128 * 32];
    __shared__ unsigned short sB[2][128 * 32];    // rows 0-63: a-half, 64-127: g-half

    const int tid   = threadIdx.x;
    const int lane  = tid & 31;
    const int wave  = tid >> 5;
    const int waveM = wave >> 1;
    const int waveN = wave & 1;
    const int m0    = blockIdx.y * 128;
    const int n0    = blockIdx.x * 64;

    const unsigned ldsA[2] = { lds_addr_of(&sA[0][0]), lds_addr_of(&sA[1][0]) };
    const unsigned ldsB[2] = { lds_addr_of(&sB[0][0]), lds_addr_of(&sB[1][0]) };

    v8f accA[2][2], accG[2][2];
    const v8f vzero = {0.f, 0.f, 0.f, 0.f, 0.f, 0.f, 0.f, 0.f};
#pragma unroll
    for (int i = 0; i < 2; ++i)
#pragma unroll
        for (int j = 0; j < 2; ++j) { accA[i][j] = vzero; accG[i][j] = vzero; }

#if HAVE_TDM
    if (wave == 0) {
        tdm_load_2d(ldsA[0], xbf + (size_t)m0 * D_, 32, 128, D_);
        tdm_load_2d(ldsB[0], WinT + (size_t)n0 * D_, 32, 64, D_);
        tdm_load_2d(ldsB[0] + 64 * 32 * 2, WinT + (size_t)(INNER_ + n0) * D_, 32, 64, D_);
    }
#endif
    for (int kt = 0; kt < D_ / 32; ++kt) {
        const int buf = kt & 1;
        const int k0  = kt * 32;
#if HAVE_TDM
        if (wave == 0) __builtin_amdgcn_s_wait_tensorcnt(0);
        __syncthreads();                       // tile kt visible to all waves
        if (kt + 1 < D_ / 32 && wave == 0) {   // overlap DMA of kt+1 with WMMAs of kt
            const int k1 = k0 + 32, nb = buf ^ 1;
            tdm_load_2d(ldsA[nb], xbf + (size_t)m0 * D_ + k1, 32, 128, D_);
            tdm_load_2d(ldsB[nb], WinT + (size_t)n0 * D_ + k1, 32, 64, D_);
            tdm_load_2d(ldsB[nb] + 64 * 32 * 2, WinT + (size_t)(INNER_ + n0) * D_ + k1, 32, 64, D_);
        }
#else
        stage_tile_bf16(sA[buf], xbf + (size_t)m0 * D_ + k0, 128, D_, tid);
        stage_tile_bf16(sB[buf], WinT + (size_t)n0 * D_ + k0, 64, D_, tid);
        stage_tile_bf16(&sB[buf][64 * 32], WinT + (size_t)(INNER_ + n0) * D_ + k0, 64, D_, tid);
        __syncthreads();
#endif
        v16bf af[2];
#pragma unroll
        for (int mt = 0; mt < 2; ++mt)
            af[mt] = load_frag(&sA[buf][(waveM * 32 + mt * 16) * 32], lane);
#pragma unroll
        for (int nt = 0; nt < 2; ++nt) {
            const v16bf bfA = load_frag(&sB[buf][(waveN * 32 + nt * 16) * 32], lane);
            const v16bf bfG = load_frag(&sB[buf][(64 + waveN * 32 + nt * 16) * 32], lane);
#pragma unroll
            for (int mt = 0; mt < 2; ++mt) {
                accA[mt][nt] = __builtin_amdgcn_wmma_f32_16x16x32_bf16(
                    false, af[mt], false, bfA, (short)0, accA[mt][nt], false, false);
                accG[mt][nt] = __builtin_amdgcn_wmma_f32_16x16x32_bf16(
                    false, af[mt], false, bfG, (short)0, accG[mt][nt], false, false);
            }
        }
        __syncthreads();                       // done with buf before it is rewritten
    }

    const int colL = lane & 15;
    const int rAdd = (lane >> 4) << 3;         // C layout: lanes 16-31 hold M = r+8
#pragma unroll
    for (int mt = 0; mt < 2; ++mt)
#pragma unroll
        for (int nt = 0; nt < 2; ++nt)
#pragma unroll
            for (int r = 0; r < 8; ++r) {
                const int gm = m0 + waveM * 32 + mt * 16 + r + rAdd;
                const int gn = n0 + waveN * 32 + nt * 16 + colL;
                const float a = accA[mt][nt][r];
                const float g = accG[mt][nt][r];
                hg[(size_t)gm * INNER_ + gn] =
                    (unsigned short)f2bf(a * (1.0f / (1.0f + __expf(-g))));
            }
}

// ---------------------------------------------------------------------------
// Kernel 2: out = silu(causal_dwconv_K4(h_gated) + bias) @ W_out
// A tile produced on the fly (hg is L2-resident, 64 of 192 MB); B tile via
// double-buffered TDM from pre-transposed bf16 W_out.
// ---------------------------------------------------------------------------
__device__ __forceinline__ void conv_acc8(float* a, uint4 hq, const float* w) {
    a[0] += bf2f(hq.x & 0xFFFFu) * w[0];
    a[1] += bf2f(hq.x >> 16)     * w[4];
    a[2] += bf2f(hq.y & 0xFFFFu) * w[8];
    a[3] += bf2f(hq.y >> 16)     * w[12];
    a[4] += bf2f(hq.z & 0xFFFFu) * w[16];
    a[5] += bf2f(hq.z >> 16)     * w[20];
    a[6] += bf2f(hq.w & 0xFFFFu) * w[24];
    a[7] += bf2f(hq.w >> 16)     * w[28];
}

__global__ __launch_bounds__(256) void conv_silu_gemm2_kernel(
        const unsigned short* __restrict__ hg,     // [M_, INNER_] bf16
        const float* __restrict__ convW,           // [INNER_, 1, KCONV]
        const float* __restrict__ convB,           // [INNER_]
        const unsigned short* __restrict__ WoutT,  // [D_, INNER_] bf16 (n-major)
        float* __restrict__ out) {                 // [M_, D_]
    __shared__ float sCW[INNER_ * KCONV];          // 32 KB
    __shared__ float sCB[INNER_];                  // 8 KB
    __shared__ unsigned short sA[128 * 32];        // conv+silu output tile
    __shared__ unsigned short sB[2][64 * 32];      // W_out^T tiles (double buffer)

    const int tid   = threadIdx.x;
    const int lane  = tid & 31;
    const int wave  = tid >> 5;
    const int waveM = wave >> 1;
    const int waveN = wave & 1;
    const int m0    = blockIdx.y * 128;
    const int n0    = blockIdx.x * 64;

    const unsigned ldsB[2] = { lds_addr_of(&sB[0][0]), lds_addr_of(&sB[1][0]) };

#pragma unroll
    for (int i = 0; i < 10; ++i) {                 // stage conv weights + bias once
        const int q = tid + i * 256;               // 0..2559 float4 segments
        if (q < 2048)       *(float4*)&sCW[q * 4] = *(const float4*)&convW[q * 4];
        else { const int b = q - 2048; *(float4*)&sCB[b * 4] = *(const float4*)&convB[b * 4]; }
    }

    v8f acc[2][2];
    const v8f vzero = {0.f, 0.f, 0.f, 0.f, 0.f, 0.f, 0.f, 0.f};
#pragma unroll
    for (int i = 0; i < 2; ++i)
#pragma unroll
        for (int j = 0; j < 2; ++j) acc[i][j] = vzero;

#if HAVE_TDM
    if (wave == 0) tdm_load_2d(ldsB[0], WoutT + (size_t)n0 * INNER_, 32, 64, INNER_);
#endif
    __syncthreads();                               // conv weights visible

    for (int kt = 0; kt < INNER_ / 32; ++kt) {
        const int buf = kt & 1;
        const int k0  = kt * 32;
        // ---- A producer: y = silu(conv(hg)+b) straight into LDS ----
#pragma unroll
        for (int i = 0; i < 2; ++i) {
            const int s   = tid + i * 256;         // 0..511
            const int row = s >> 2;
            const int cg  = (s & 3) << 3;          // 8-channel group
            const int f   = k0 + cg;
            const int m   = m0 + row;
            const int l   = m & (L_ - 1);
            float a[8];
#pragma unroll
            for (int c = 0; c < 8; ++c) a[c] = sCB[f + c];
#pragma unroll
            for (int j = 0; j < KCONV; ++j) {
                if (l + j >= KCONV - 1) {          // causal left pad
                    const uint4 hq =
                        *(const uint4*)&hg[(size_t)(m - (KCONV - 1) + j) * INNER_ + f];
                    conv_acc8(a, hq, &sCW[f * KCONV + j]);
                }
            }
            unsigned p[4];
#pragma unroll
            for (int c = 0; c < 4; ++c) {
                const float v0 = a[2 * c],     y0 = v0 * (1.0f / (1.0f + __expf(-v0)));
                const float v1 = a[2 * c + 1], y1 = v1 * (1.0f / (1.0f + __expf(-v1)));
                p[c] = f2bf(y0) | (f2bf(y1) << 16);
            }
            *(uint4*)&sA[row * 32 + cg] = make_uint4(p[0], p[1], p[2], p[3]);
        }
        if (kt + 1 < INNER_ / 32)
            __builtin_prefetch(&hg[(size_t)(m0 + (tid >> 1)) * INNER_ + k0 + 32], 0, 3);
#if HAVE_TDM
        if (wave == 0) __builtin_amdgcn_s_wait_tensorcnt(0);
        __syncthreads();
        if (kt + 1 < INNER_ / 32 && wave == 0)
            tdm_load_2d(ldsB[buf ^ 1], WoutT + (size_t)n0 * INNER_ + k0 + 32, 32, 64, INNER_);
#else
        stage_tile_bf16(sB[buf], WoutT + (size_t)n0 * INNER_ + k0, 64, INNER_, tid);
        __syncthreads();
#endif
        v16bf af[2];
#pragma unroll
        for (int mt = 0; mt < 2; ++mt)
            af[mt] = load_frag(&sA[(waveM * 32 + mt * 16) * 32], lane);
#pragma unroll
        for (int nt = 0; nt < 2; ++nt) {
            const v16bf bfr = load_frag(&sB[buf][(waveN * 32 + nt * 16) * 32], lane);
#pragma unroll
            for (int mt = 0; mt < 2; ++mt)
                acc[mt][nt] = __builtin_amdgcn_wmma_f32_16x16x32_bf16(
                    false, af[mt], false, bfr, (short)0, acc[mt][nt], false, false);
        }
        __syncthreads();
    }

    const int colL = lane & 15;
    const int rAdd = (lane >> 4) << 3;
#pragma unroll
    for (int mt = 0; mt < 2; ++mt)
#pragma unroll
        for (int nt = 0; nt < 2; ++nt)
#pragma unroll
            for (int r = 0; r < 8; ++r) {
                const int gm = m0 + waveM * 32 + mt * 16 + r + rAdd;
                const int gn = n0 + waveN * 32 + nt * 16 + colL;
                out[(size_t)gm * D_ + gn] = acc[mt][nt][r];
            }
}

// ---------------------------------------------------------------------------
extern "C" void kernel_launch(void* const* d_in, const int* in_sizes, int n_in,
                              void* d_out, int out_size, void* d_ws, size_t ws_size,
                              hipStream_t stream) {
    (void)in_sizes; (void)n_in; (void)out_size; (void)ws_size;
    const float* x     = (const float*)d_in[0];
    const float* W_in  = (const float*)d_in[1];
    const float* convW = (const float*)d_in[2];
    const float* convB = (const float*)d_in[3];
    const float* W_out = (const float*)d_in[4];
    float* out = (float*)d_out;

    // workspace layout (bytes)
    char* ws = (char*)d_ws;
    unsigned short* hg    = (unsigned short*)(ws);                         // 64 MB
    unsigned short* xbf   = (unsigned short*)(ws + (size_t)67108864);      // 32 MB
    unsigned short* WinT  = (unsigned short*)(ws + (size_t)100663296);     //  8 MB
    unsigned short* WoutT = (unsigned short*)(ws + (size_t)109051904);     //  4 MB

    // preprocessing: bf16 conversion + K-contiguous weight layouts
    cvt_bf16_kernel<<<dim3((M_ * D_ / 8) / 256), dim3(256), 0, stream>>>(x, xbf, M_ * D_ / 8);
    transpose_bf16_kernel<<<dim3(2 * INNER_ / 64, D_ / 64), dim3(256), 0, stream>>>(
        W_in, WinT, D_, 2 * INNER_);
    transpose_bf16_kernel<<<dim3(D_ / 64, INNER_ / 64), dim3(256), 0, stream>>>(
        W_out, WoutT, INNER_, D_);

    gemm1_glu_kernel<<<dim3(INNER_ / 64, M_ / 128), dim3(256), 0, stream>>>(xbf, WinT, hg);
    conv_silu_gemm2_kernel<<<dim3(D_ / 64, M_ / 128), dim3(256), 0, stream>>>(
        hg, convW, convB, WoutT, out);
}
